// MFBasedModel_66529043415381
// MI455X (gfx1250) — matrix operand order repaired and checked by reading, not compiled, not run on hardware
//
#include <hip/hip_runtime.h>
#include <hip/hip_bf16.h>

// ---------- problem constants ----------
#define BN    16384
#define CNUM  2048
#define KTOP  8
#define DDIM  10
#define MDIM  50
#define KPAD  64       // padded K for bf16 WMMA (50 -> 64, 60 -> 64)

// base-2 domain constants: y = z * log2(e) / tau
#define LN2F      0.69314718055994531f
#define Y_SCALE   0.14426950408889634f   // log2(e) / tau, tau = 10

// ---------- WMMA types ----------
typedef __attribute__((ext_vector_type(16))) __bf16 v16bf;
typedef __attribute__((ext_vector_type(8)))  float  v8f;

union BF16x16 { v16bf v; uint4 q[2]; };

#define WMMA_BF16(A,B,C) \
  __builtin_amdgcn_wmma_f32_16x16x32_bf16(false,(A),false,(B),(short)0,(C),false,false)

static __device__ __forceinline__ float fast_log2(float x) { return __builtin_amdgcn_logf(x); }
static __device__ __forceinline__ float fast_exp2(float x) { return __builtin_amdgcn_exp2f(x); }

static __device__ __forceinline__ unsigned short f2bf(float x) {
  unsigned int u = __float_as_uint(x);
  unsigned int r = (u + 0x7FFFu + ((u >> 16) & 1u)) >> 16;   // RNE
  return (unsigned short)r;
}

// A-operand (16x32 bf16 tile) per CDNA5 layout: lane<16 holds K=[kb..kb+8)+[kb+16..kb+24),
// lane>=16 holds K=[kb+8..kb+16)+[kb+24..kb+32) of row (lane&15).
static __device__ __forceinline__ v16bf load_a_bf16(const unsigned short* buf, int ld,
                                                    int rowbase, int kbase, int lane) {
  int hi = lane >> 4;
  const unsigned short* p = buf + (size_t)(rowbase + (lane & 15)) * ld + kbase + hi * 8;
  BF16x16 u;
  u.q[0] = *(const uint4*)(p);
  u.q[1] = *(const uint4*)(p + 16);
  return u.v;
}

// B-operand (32x16 bf16 tile) from an N-major buffer buf[n][ld]: lane n<16 holds
// K=[kb..kb+8)+[kb+16..kb+24) of column n; lane>=16 holds K=[kb+8..kb+16)+[kb+24..kb+32) of column n-16.
static __device__ __forceinline__ v16bf load_b_bf16(const unsigned short* buf, int ld,
                                                    int nbase, int kbase, int lane) {
  int hi = lane >> 4;
  const unsigned short* p = buf + (size_t)(nbase + (lane & 15)) * ld + kbase + hi * 8;
  BF16x16 u;
  u.q[0] = *(const uint4*)(p);
  u.q[1] = *(const uint4*)(p + 16);
  return u.v;
}

// ---------- weight prep: bf16 pad/transpose ----------
__global__ __launch_bounds__(256) void k_wprep(const float* __restrict__ C,
                                               const float* __restrict__ mw1,
                                               const float* __restrict__ mw2,
                                               unsigned short* __restrict__ Cbf,
                                               unsigned short* __restrict__ W1t,
                                               unsigned short* __restrict__ W2t) {
  int t = blockIdx.x * 256 + threadIdx.x;
  if (t < CNUM * KPAD) {
    int c = t >> 6, k = t & 63;
    Cbf[t] = f2bf(k < MDIM ? C[c * MDIM + k] : 0.f);
  } else if (t < CNUM * KPAD + 64 * 64) {
    int t2 = t - CNUM * KPAD;
    int n = t2 >> 6, k = t2 & 63;                  // W1t[n][k] = mw1[k][n]
    W1t[t2] = f2bf((k < 60 && n < 50) ? mw1[k * 50 + n] : 0.f);
  } else if (t < CNUM * KPAD + 64 * 64 + 112 * 64) {
    int t3 = t - CNUM * KPAD - 64 * 64;
    int n = t3 >> 6, k = t3 & 63;                  // W2t[n][k] = mw2[k][n]
    W2t[t3] = f2bf((k < 50 && n < 100) ? mw2[k * 100 + n] : 0.f);
  }
}

// ---------- gather + uid@w1 projection, bf16 K-padded ----------
__global__ __launch_bounds__(128) void k_prep(const int* __restrict__ x,
                                              const float* __restrict__ srcT,
                                              const float* __restrict__ tgtT,
                                              const float* __restrict__ w1,
                                              unsigned short* __restrict__ Upad,
                                              unsigned short* __restrict__ Tpad,
                                              float* __restrict__ uidf) {
  __shared__ float w1s[DDIM * MDIM];
  for (int i = threadIdx.x; i < DDIM * MDIM; i += 128) w1s[i] = w1[i];
  __syncthreads();
  int b = blockIdx.x * 128 + threadIdx.x;
  int u = x[2 * b];
  float uid[DDIM], tuid[DDIM];
#pragma unroll
  for (int d = 0; d < DDIM; ++d) {
    uid[d]  = srcT[(size_t)u * DDIM + d];
    tuid[d] = tgtT[(size_t)u * DDIM + d];
    uidf[(size_t)b * DDIM + d] = uid[d];
  }
  for (int m = 0; m < KPAD; ++m) {
    float su = 0.f, st = 0.f;
    if (m < MDIM) {
#pragma unroll
      for (int d = 0; d < DDIM; ++d) {
        float wv = w1s[d * MDIM + m];
        su += uid[d] * wv;
        st += tuid[d] * wv;
      }
    }
    Upad[(size_t)b * KPAD + m] = f2bf(su);
    Tpad[(size_t)b * KPAD + m] = f2bf(st);
  }
}

// ---------- routing: bf16 WMMA GEMM + online softmax (base-2) + top-8 ----------
#define RW 8   // waves per block
__global__ __launch_bounds__(256) void k_route(const unsigned short* __restrict__ Apad,
                                               const unsigned short* __restrict__ Cbf,
                                               const float* __restrict__ gum,
                                               float* __restrict__ topw,
                                               int*   __restrict__ topi) {
  __shared__ float zm[RW][16][36];
  const int wave = threadIdx.x >> 5, lane = threadIdx.x & 31;
  const int hi = lane >> 4, lcol = lane & 15;
  const int tile = (blockIdx.x * RW + wave) * 16;

  // A operands live in registers across the whole cluster loop.
  v16bf a0 = load_a_bf16(Apad, KPAD, tile, 0, lane);
  v16bf a1 = load_a_bf16(Apad, KPAD, tile, 32, lane);

  float tv[KTOP]; int tix[KTOP];
#pragma unroll
  for (int i = 0; i < KTOP; ++i) { tv[i] = -1e30f; tix[i] = 0; }
  float mx = -1e30f, sm = 0.f;
  const int cbase = hi * 8;          // this lane scans cols [cbase, cbase+8) of row lcol

  for (int cb = 0; cb < CNUM; cb += 16) {
    v16bf b0 = load_b_bf16(Cbf, KPAD, cb, 0, lane);
    v16bf b1 = load_b_bf16(Cbf, KPAD, cb, 32, lane);
    v8f acc = {0.f, 0.f, 0.f, 0.f, 0.f, 0.f, 0.f, 0.f};
    acc = WMMA_BF16(a0, b0, acc);
    acc = WMMA_BF16(a1, b1, acc);

    __builtin_prefetch(&gum[(size_t)(tile + lcol) * CNUM + cb + 128], 0, 1);

    // gumbel in base-2: g = -ln2*log2(-ln2*log2(u+eps) + eps); y = (s+g)*log2e/tau
#pragma unroll
    for (int r = 0; r < 8; ++r) {
      int rowg = tile + r + hi * 8;
      float u = gum[(size_t)rowg * CNUM + cb + lcol];
      float l1 = fast_log2(u + 1e-20f);
      float inner = fmaf(-LN2F, l1, 1e-20f);
      float g = -LN2F * fast_log2(inner);
      zm[wave][r + hi * 8][lcol] = (acc[r] + g) * Y_SCALE;
    }
    __syncthreads();

    // online softmax (base 2) + register top-8 over this lane's half-row
#pragma unroll
    for (int j = 0; j < 8; ++j) {
      float y = zm[wave][lcol][cbase + j];
      int ci = cb + cbase + j;
      if (y > mx) { sm = sm * fast_exp2(mx - y) + 1.f; mx = y; }
      else        { sm += fast_exp2(y - mx); }
      if (y > tv[KTOP - 1]) {
        float cv = y; int cc = ci;
#pragma unroll
        for (int i = 0; i < KTOP; ++i) {
          if (cv > tv[i]) { float t0 = tv[i]; int t1 = tix[i]; tv[i] = cv; tix[i] = cc; cv = t0; cc = t1; }
        }
      }
    }
    __syncthreads();
  }

  // publish half-row results, merge the lane pair through LDS
  int off = hi * 16;
#pragma unroll
  for (int i = 0; i < KTOP; ++i) {
    zm[wave][lcol][off + i]     = tv[i];
    zm[wave][lcol][off + 8 + i] = __int_as_float(tix[i]);
  }
  zm[wave][lcol][32 + hi * 2] = mx;
  zm[wave][lcol][33 + hi * 2] = sm;
  __syncthreads();

  if (lane < 16) {
    float mxl = zm[wave][lcol][32], sml = zm[wave][lcol][33];
    float mxh = zm[wave][lcol][34], smh = zm[wave][lcol][35];
    float M = fmaxf(mxl, mxh);
    float invS = 1.f / (sml * fast_exp2(mxl - M) + smh * fast_exp2(mxh - M));
    int ia = 0, ib = 0;
    size_t base = (size_t)(tile + lcol) * KTOP;
#pragma unroll
    for (int k = 0; k < KTOP; ++k) {
      float av = zm[wave][lcol][ia];
      float bv = zm[wave][lcol][16 + ib];
      bool ta = av >= bv;
      float v = ta ? av : bv;
      int  id = __float_as_int(ta ? zm[wave][lcol][8 + ia] : zm[wave][lcol][24 + ib]);
      ia += ta ? 1 : 0; ib += ta ? 0 : 1;
      topw[base + k] = fast_exp2(v - M) * invS;
      topi[base + k] = id;
    }
  }
}

// ---------- fused per-(b,k) MLP with WMMA + epilogue reduce ----------
#define TW 4   // waves per block
__global__ __launch_bounds__(128) void k_tail(const unsigned short* __restrict__ Cbf,
                                              const unsigned short* __restrict__ W1t,
                                              const unsigned short* __restrict__ W2t,
                                              const float* __restrict__ mb1,
                                              const float* __restrict__ mb2,
                                              const float* __restrict__ uidf,
                                              const float* __restrict__ topw_s,
                                              const int*   __restrict__ topi_s,
                                              float* __restrict__ pre) {
  __shared__ unsigned short w1s[64 * 64];
  __shared__ unsigned short w2s[112 * 64];
  __shared__ float b1s[64], b2s[112];
  __shared__ __align__(16) float scr[TW][16 * 113 + 12];   // wave-private scratch (X/H/Map alias)

  for (int i = threadIdx.x; i < 64 * 64 / 2;  i += 128) ((unsigned int*)w1s)[i] = ((const unsigned int*)W1t)[i];
  for (int i = threadIdx.x; i < 112 * 64 / 2; i += 128) ((unsigned int*)w2s)[i] = ((const unsigned int*)W2t)[i];
  for (int i = threadIdx.x; i < 64;  i += 128) b1s[i] = (i < 50)  ? mb1[i] : 0.f;
  for (int i = threadIdx.x; i < 112; i += 128) b2s[i] = (i < 100) ? mb2[i] : 0.f;
  __syncthreads();

  const int wave = threadIdx.x >> 5, lane = threadIdx.x & 31;
  const int hi = lane >> 4, lcol = lane & 15;
  const int ptile = (blockIdx.x * TW + wave) * 16;
  unsigned short* xs = (unsigned short*)scr[wave];          // 16x64 bf16 input tile

  // build X row = [ C[topi_s[b][k]] (50) | uid[b] (10) | 0 pad (4) ], two lanes per row
  {
    int p = ptile + lcol, b = p >> 3, k = p & 7;
    int cidx = topi_s[b * KTOP + k];
    const unsigned int* crow = (const unsigned int*)(Cbf + (size_t)cidx * KPAD);
    unsigned int* xr = (unsigned int*)(xs + lcol * KPAD);
    if (hi == 0) {
#pragma unroll
      for (int j = 0; j < 16; ++j) xr[j] = crow[j];
    } else {
#pragma unroll
      for (int j = 16; j < 25; ++j) xr[j] = crow[j];
      const float* uf = uidf + (size_t)b * DDIM;
#pragma unroll
      for (int j = 0; j < 5; ++j) {
        unsigned int lo = f2bf(uf[2 * j]);
        unsigned int hi16 = f2bf(uf[2 * j + 1]);
        xr[25 + j] = lo | (hi16 << 16);
      }
      xr[30] = 0u; xr[31] = 0u;
    }
  }
  __syncthreads();
  v16bf a0 = load_a_bf16(xs, KPAD, 0, 0, lane);
  v16bf a1 = load_a_bf16(xs, KPAD, 0, 32, lane);
  __syncthreads();

  // GEMM1: H = relu(X @ mw1 + mb1), write bf16 (aliases xs; X already in regs)
  unsigned short* hs = xs;
#pragma unroll
  for (int nt = 0; nt < 4; ++nt) {
    v16bf b0 = load_b_bf16(w1s, KPAD, nt * 16, 0, lane);
    v16bf b1 = load_b_bf16(w1s, KPAD, nt * 16, 32, lane);
    v8f acc = {0.f, 0.f, 0.f, 0.f, 0.f, 0.f, 0.f, 0.f};
    acc = WMMA_BF16(a0, b0, acc);
    acc = WMMA_BF16(a1, b1, acc);
    int n = nt * 16 + lcol;
    float bias = b1s[n];
#pragma unroll
    for (int r = 0; r < 8; ++r)
      hs[(r + hi * 8) * KPAD + n] = f2bf(fmaxf(acc[r] + bias, 0.f));
  }
  __syncthreads();
  v16bf a20 = load_a_bf16(hs, KPAD, 0, 0, lane);
  v16bf a21 = load_a_bf16(hs, KPAD, 0, 32, lane);
  __syncthreads();

  // GEMM2: Map = H @ mw2 + mb2 (f32 tile in LDS, aliases hs)
  float* ms = scr[wave];
#pragma unroll
  for (int nt = 0; nt < 7; ++nt) {
    v16bf b0 = load_b_bf16(w2s, KPAD, nt * 16, 0, lane);
    v16bf b1 = load_b_bf16(w2s, KPAD, nt * 16, 32, lane);
    v8f acc = {0.f, 0.f, 0.f, 0.f, 0.f, 0.f, 0.f, 0.f};
    acc = WMMA_BF16(a20, b0, acc);
    acc = WMMA_BF16(a21, b1, acc);
    int n = nt * 16 + lcol;
    float bias = b2s[n];
#pragma unroll
    for (int r = 0; r < 8; ++r) ms[(r + hi * 8) * 113 + n] = acc[r] + bias;
  }
  __syncthreads();

  // epilogue: se[e] = sum_d uid[d]*Map[d*10+e]; reduce tw*se over the 8 k's of each b
  if (lane < 16) {
    int p = ptile + lane, b = p >> 3, k = p & 7;
    float tw = topw_s[b * KTOP + k];
    const float* uf = uidf + (size_t)b * DDIM;
    float ud[DDIM];
#pragma unroll
    for (int d = 0; d < DDIM; ++d) ud[d] = uf[d];
    float ve[DDIM];
#pragma unroll
    for (int e = 0; e < DDIM; ++e) {
      float s = 0.f;
#pragma unroll
      for (int d = 0; d < DDIM; ++d) s += ud[d] * ms[lane * 113 + d * DDIM + e];
      ve[e] = tw * s;
    }
#pragma unroll
    for (int e = 0; e < DDIM; ++e) {
      ve[e] += __shfl_down(ve[e], 4);
      ve[e] += __shfl_down(ve[e], 2);
      ve[e] += __shfl_down(ve[e], 1);
    }
    if ((lane & 7) == 0) {
      float* pb = pre + (size_t)b * DDIM;
#pragma unroll
      for (int e = 0; e < DDIM; ++e) pb[e] = ve[e];
    }
  }
}

// ---------- final: tgt combine @ w2 + pre, dot with iid embedding ----------
__global__ __launch_bounds__(256) void k_final(const int* __restrict__ x,
                                               const float* __restrict__ tgt_iid,
                                               const float* __restrict__ C,
                                               const float* __restrict__ w2,
                                               const float* __restrict__ topw_t,
                                               const int*   __restrict__ topi_t,
                                               const float* __restrict__ pre,
                                               float* __restrict__ out) {
  int b = blockIdx.x * 256 + threadIdx.x;
  float comb[MDIM];
#pragma unroll
  for (int m = 0; m < MDIM; ++m) comb[m] = 0.f;
#pragma unroll
  for (int k = 0; k < KTOP; ++k) {
    float w = topw_t[(size_t)b * KTOP + k];
    const float* cr = C + (size_t)topi_t[(size_t)b * KTOP + k] * MDIM;
    for (int m = 0; m < MDIM; ++m) comb[m] += w * cr[m];
  }
  int ii = x[2 * b + 1];
  const float* ie = tgt_iid + (size_t)ii * DDIM;
  const float* pb = pre + (size_t)b * DDIM;
  float o = 0.f;
#pragma unroll
  for (int e = 0; e < DDIM; ++e) {
    float t = 0.f;
    for (int m = 0; m < MDIM; ++m) t += comb[m] * w2[m * DDIM + e];
    o += (t + pb[e]) * ie[e];
  }
  out[b] = o;
}

// ---------- launch ----------
extern "C" void kernel_launch(void* const* d_in, const int* in_sizes, int n_in,
                              void* d_out, int out_size, void* d_ws, size_t ws_size,
                              hipStream_t stream) {
  const int*   x      = (const int*)  d_in[0];
  const float* gum_s  = (const float*)d_in[1];
  const float* gum_t  = (const float*)d_in[2];
  const float* srcT   = (const float*)d_in[3];
  const float* tgtT   = (const float*)d_in[4];
  const float* iidT   = (const float*)d_in[5];
  const float* C      = (const float*)d_in[6];
  const float* w1     = (const float*)d_in[7];
  const float* w2     = (const float*)d_in[8];
  const float* mw1    = (const float*)d_in[9];
  const float* mb1    = (const float*)d_in[10];
  const float* mw2    = (const float*)d_in[11];
  const float* mb2    = (const float*)d_in[12];
  float* out = (float*)d_out;

  char* w = (char*)d_ws;
  unsigned short* Upad = (unsigned short*)w; w += (size_t)BN * KPAD * 2;
  unsigned short* Tpad = (unsigned short*)w; w += (size_t)BN * KPAD * 2;
  unsigned short* Cbf  = (unsigned short*)w; w += (size_t)CNUM * KPAD * 2;
  unsigned short* W1t  = (unsigned short*)w; w += 64 * 64 * 2;
  unsigned short* W2t  = (unsigned short*)w; w += 112 * 64 * 2;
  float* uidf   = (float*)w; w += (size_t)BN * DDIM * 4;
  float* topw_s = (float*)w; w += (size_t)BN * KTOP * 4;
  int*   topi_s = (int*)  w; w += (size_t)BN * KTOP * 4;
  float* topw_t = (float*)w; w += (size_t)BN * KTOP * 4;
  int*   topi_t = (int*)  w; w += (size_t)BN * KTOP * 4;
  float* pre    = (float*)w; w += (size_t)BN * DDIM * 4;

  (void)in_sizes; (void)n_in; (void)out_size; (void)ws_size;

  int wtot = CNUM * KPAD + 64 * 64 + 112 * 64;
  k_wprep<<<(wtot + 255) / 256, 256, 0, stream>>>(C, mw1, mw2, Cbf, W1t, W2t);
  k_prep<<<BN / 128, 128, 0, stream>>>(x, srcT, tgtT, w1, Upad, Tpad, uidf);
  k_route<<<BN / (16 * RW), 256, 0, stream>>>(Upad, Cbf, gum_s, topw_s, topi_s);
  k_route<<<BN / (16 * RW), 256, 0, stream>>>(Tpad, Cbf, gum_t, topw_t, topi_t);
  k_tail<<<(BN * KTOP) / (16 * TW), 128, 0, stream>>>(Cbf, W1t, W2t, mb1, mb2,
                                                      uidf, topw_s, topi_s, pre);
  k_final<<<BN / 256, 256, 0, stream>>>(x, iidT, C, w2, topw_t, topi_t, pre, out);
}
